// ScaledDotProductAttention_16114717295168
// MI455X (gfx1250) — matrix operand order repaired and checked by reading, not compiled, not run on hardware
//
#include <hip/hip_runtime.h>

// Problem constants (reference: B=64, L=512, D=64, all f32)
constexpr int B_ = 64;
constexpr int L_ = 512;
constexpr int D_ = 64;

typedef __attribute__((ext_vector_type(16))) _Float16 v16h;
typedef __attribute__((ext_vector_type(8)))  _Float16 v8h;
typedef __attribute__((ext_vector_type(8)))  float    v8f;

// ---------------------------------------------------------------------------
// WMMA operand loaders (f16 sources), matching CDNA5 wave32 16-bit layouts
// (CDNA5 ISA §7.12.2):
//  A 16x32: lane L holds row m=L%16; lanes 0-15 K={0..7,16..23},
//           lanes 16-31 K={8..15,24..31}.
//  B 32x16: lane L holds col n=L%16; lanes 0-15 K={0..7,16..23},
//           lanes 16-31 K={8..15,24..31}.
//  C/D 16x16 f32: lane L col n=L%16, VGPR r -> row m=r+(lane<16?0:8).
// ---------------------------------------------------------------------------
__device__ __forceinline__ v16h wmma_a_f16(const _Float16* base, int ldRow) {
  const int lane = threadIdx.x & 31;
  const int m    = lane & 15;
  const int kb   = (lane < 16) ? 0 : 8;
  const _Float16* p = base + (size_t)m * ldRow + kb;
  const v8h lo = *(const v8h*)p;        // K = kb..kb+7
  const v8h hi = *(const v8h*)(p + 16); // K = 16+kb..16+kb+7
  v16h a;
#pragma unroll
  for (int i = 0; i < 8; ++i) { a[i] = lo[i]; a[8 + i] = hi[i]; }
  return a;
}

__device__ __forceinline__ v16h wmma_b_f16(const _Float16* base, int strideN) {
  const int lane = threadIdx.x & 31;
  const int n    = lane & 15;
  const int kb   = (lane < 16) ? 0 : 8;
  const _Float16* p = base + (size_t)n * strideN + kb; // K contiguous
  const v8h lo = *(const v8h*)p;
  const v8h hi = *(const v8h*)(p + 16);
  v16h b;
#pragma unroll
  for (int i = 0; i < 8; ++i) { b[i] = lo[i]; b[8 + i] = hi[i]; }
  return b;
}

__device__ __forceinline__ v8f wmma32(const v16h& a, const v16h& b, const v8f& c) {
  return __builtin_amdgcn_wmma_f32_16x16x32_f16(false, a, false, b, (short)0, c,
                                                false, false);
}

// ---------------------------------------------------------------------------
// K0a: one-shot f32 -> f16 convert (RNE), 8 elements/thread, b128 in/out.
// ---------------------------------------------------------------------------
__global__ void __launch_bounds__(256)
cvt_kernel(const float* __restrict__ src, _Float16* __restrict__ dst, int n) {
  const int idx = (blockIdx.x * 256 + threadIdx.x) * 8;
  if (idx + 8 > n) return;
  const float4 f0 = *(const float4*)(src + idx);
  const float4 f1 = *(const float4*)(src + idx + 4);
  v8h h;
  h[0] = (_Float16)f0.x; h[1] = (_Float16)f0.y;
  h[2] = (_Float16)f0.z; h[3] = (_Float16)f0.w;
  h[4] = (_Float16)f1.x; h[5] = (_Float16)f1.y;
  h[6] = (_Float16)f1.z; h[7] = (_Float16)f1.w;
  *(v8h*)(dst + idx) = h;
}

// ---------------------------------------------------------------------------
// K0b: per-matrix transpose + convert: src m: [rows][cols] f32 ->
//      dst m: [cols][rows] f16.  rows=512, cols=64; LDS 64x64 tile (+1 pad).
// Grid: (rows/64, nmat), block = 256.
// ---------------------------------------------------------------------------
__global__ void __launch_bounds__(256)
transpose_cvt_kernel(const float* __restrict__ src, _Float16* __restrict__ dst,
                     int rows, int cols) {
  __shared__ float t[64][65];
  const int r0 = blockIdx.x * 64;
  const int m  = blockIdx.y;
  const float* s = src + (size_t)m * rows * cols;
  _Float16*    d = dst + (size_t)m * rows * cols;

  const int tr = threadIdx.x >> 2;        // 0..63
  const int tc = (threadIdx.x & 3) * 16;  // 0,16,32,48
  const float* sp = s + (size_t)(r0 + tr) * cols + tc;
#pragma unroll
  for (int j = 0; j < 16; j += 4) {
    const float4 f = *(const float4*)(sp + j);
    t[tr][tc + j]     = f.x; t[tr][tc + j + 1] = f.y;
    t[tr][tc + j + 2] = f.z; t[tr][tc + j + 3] = f.w;
  }
  __syncthreads();

  v8h h0, h1;
#pragma unroll
  for (int j = 0; j < 8; ++j) h0[j] = (_Float16)t[tc + j][tr];
#pragma unroll
  for (int j = 0; j < 8; ++j) h1[j] = (_Float16)t[tc + 8 + j][tr];
  _Float16* dp = d + (size_t)tr * rows + r0 + tc;
  *(v8h*)dp       = h0;
  *(v8h*)(dp + 8) = h1;
}

// ---------------------------------------------------------------------------
// K1: bias[b][kj][qi] = sum_d edge_key[kj][qi][d] * q[b][kj][d]
// Per kj: GEMM A[b][d]=qh[:,kj,:] x B[d][qi]=ekh[kj]^T.  Stored [b][kj][qi].
// All 8 B-tiles loaded into distinct registers first -> pipelined loads.
// Grid: (L, B/16), block = 256 (8 waves); wave w covers qi n0 = w*64.
// ---------------------------------------------------------------------------
__global__ void __launch_bounds__(256)
bias_kernel(const _Float16* __restrict__ qh, const _Float16* __restrict__ ekh,
            float* __restrict__ bias) {
  const int kj   = blockIdx.x;
  const int b0   = blockIdx.y * 16;
  const int wave = threadIdx.x >> 5;
  const int lane = threadIdx.x & 31;

  const _Float16* aBase = qh + ((size_t)b0 * L_ + kj) * D_;
  const v16h a0 = wmma_a_f16(aBase, L_ * D_);
  const v16h a1 = wmma_a_f16(aBase + 32, L_ * D_);

  v16h bt[8];
#pragma unroll
  for (int nt = 0; nt < 4; ++nt) {
    const _Float16* bBase =
        ekh + ((size_t)kj * L_ + wave * 64 + nt * 16) * D_; // B[d][qi]
    bt[2 * nt]     = wmma_b_f16(bBase, D_);
    bt[2 * nt + 1] = wmma_b_f16(bBase + 32, D_);
  }

  v8f c[4] = {};
#pragma unroll
  for (int nt = 0; nt < 4; ++nt) {
    c[nt] = wmma32(a0, bt[2 * nt], c[nt]);
    c[nt] = wmma32(a1, bt[2 * nt + 1], c[nt]);
  }

  const int n  = lane & 15;
  const int mo = (lane < 16) ? 0 : 8;
#pragma unroll
  for (int nt = 0; nt < 4; ++nt) {
    float* dst =
        bias + (((size_t)(b0 + mo) * L_ + kj) * L_) + wave * 64 + nt * 16 + n;
#pragma unroll
    for (int r = 0; r < 8; ++r) dst[(size_t)r * L_ * L_] = c[nt][r];
  }
}

// ---------------------------------------------------------------------------
// K2: fused attention for one (batch b, 16-query tile).
//   scores = Q K^T -> LDS [16][512]; += bias; *scale; mask -> -inf; softmax;
//   P (f16) -> ws; out = P V -> d_out.   Block = 1 wave. Grid: (B, L/16).
// GEMM loops software double-buffered (next tile's loads issued before the
// current tile's WMMAs) so loads overlap matrix-core execution in-wave.
// ---------------------------------------------------------------------------
__global__ void __launch_bounds__(32)
attn_kernel(const _Float16* __restrict__ qh, const _Float16* __restrict__ kh,
            const _Float16* __restrict__ vT, const float* __restrict__ scale_p,
            const unsigned char* __restrict__ mask,
            const float* __restrict__ bias, _Float16* __restrict__ P,
            float* __restrict__ out) {
  const int b    = blockIdx.x;
  const int q0   = blockIdx.y * 16;
  const int lane = threadIdx.x & 31;
  const float scale = *scale_p;
  const int n  = lane & 15;
  const int mo = (lane < 16) ? 0 : 8;

  __shared__ float sc[16][L_ + 4]; // +4 pad: avoid bank conflicts on columns

  // ---- Phase 1: raw content scores Q K^T (double-buffered B tiles) ----
  const _Float16* aBase = qh + ((size_t)b * L_ + q0) * D_;
  const v16h a0 = wmma_a_f16(aBase, D_);
  const v16h a1 = wmma_a_f16(aBase + 32, D_);

  const _Float16* kbase = kh + (size_t)b * L_ * D_; // B[d][kj] per 16-kj tile
  v16h nb0 = wmma_b_f16(kbase, D_);
  v16h nb1 = wmma_b_f16(kbase + 32, D_);
  for (int kt = 0; kt < L_ / 16; ++kt) {
    const v16h cb0 = nb0, cb1 = nb1;
    if (kt + 1 < L_ / 16) {
      const _Float16* bB = kbase + (size_t)(kt + 1) * 16 * D_;
      nb0 = wmma_b_f16(bB, D_);
      nb1 = wmma_b_f16(bB + 32, D_);
    }
    v8f c = {};
    c = wmma32(a0, cb0, c);
    c = wmma32(a1, cb1, c);
#pragma unroll
    for (int r = 0; r < 8; ++r) sc[mo + r][kt * 16 + n] = c[r];
  }
  __syncthreads();

  // ---- Phase 2a: add bias (layout [b][kj][qi]) and scale ----
  const float* bb = bias + (size_t)b * L_ * L_ + q0;
  for (int t = 0; t < 16; ++t) {
    const int kj = t * 32 + lane;
    const float* bp = bb + (size_t)kj * L_;
    float breg[16];
#pragma unroll
    for (int qi = 0; qi < 16; ++qi) breg[qi] = bp[qi];
#pragma unroll
    for (int qi = 0; qi < 16; ++qi)
      sc[qi][kj] = (sc[qi][kj] + breg[qi]) * scale;
  }
  __syncthreads();

  // ---- Phase 2b: mask (True -> -inf); 16 bytes/lane coalesced ----
  const unsigned char* mp = mask + ((size_t)b * L_ + q0) * L_;
  for (int qi = 0; qi < 16; ++qi) {
    const uint4 mw = *(const uint4*)(mp + (size_t)qi * L_ + lane * 16);
    const unsigned char* mb = (const unsigned char*)&mw;
#pragma unroll
    for (int j = 0; j < 16; ++j)
      if (mb[j]) sc[qi][lane * 16 + j] = -__builtin_inff();
  }
  __syncthreads();

  // ---- Phase 3: row softmax (wave-cooperative, shuffle reductions) ----
  for (int qi = 0; qi < 16; ++qi) {
    float mx = -__builtin_inff();
    for (int t = 0; t < 16; ++t) mx = fmaxf(mx, sc[qi][t * 32 + lane]);
#pragma unroll
    for (int off = 16; off > 0; off >>= 1) mx = fmaxf(mx, __shfl_xor(mx, off, 32));
    float sum = 0.f;
    for (int t = 0; t < 16; ++t) {
      const float e = __expf(sc[qi][t * 32 + lane] - mx);
      sc[qi][t * 32 + lane] = e;
      sum += e;
    }
#pragma unroll
    for (int off = 16; off > 0; off >>= 1) sum += __shfl_xor(sum, off, 32);
    const float inv = 1.0f / sum;
    for (int t = 0; t < 16; ++t) sc[qi][t * 32 + lane] *= inv;
  }
  __syncthreads();

  // ---- Phase 4: write P (f16) for K3 ----
  _Float16* pw = P + ((size_t)b * L_ + q0) * L_;
  for (int qi = 0; qi < 16; ++qi) {
    const float* srow = &sc[qi][lane * 16];
    v8h h0, h1;
#pragma unroll
    for (int j = 0; j < 8; ++j) h0[j] = (_Float16)srow[j];
#pragma unroll
    for (int j = 0; j < 8; ++j) h1[j] = (_Float16)srow[8 + j];
    _Float16* dp = pw + (size_t)qi * L_ + lane * 16;
    *(v8h*)dp       = h0;
    *(v8h*)(dp + 8) = h1;
  }

  // ---- Phase 5: out = P V  (16x64, K=512; B from vT[b][d][kj]) ----
  const _Float16* vbase = vT + (size_t)b * D_ * L_;
  const int kbA = (lane < 16) ? 0 : 8;
  auto loadA = [&](int kt) {
    const float* p = &sc[lane & 15][kt * 32 + kbA];
    v16h a;
#pragma unroll
    for (int i = 0; i < 8; ++i) a[i] = (_Float16)p[i];
#pragma unroll
    for (int i = 0; i < 8; ++i) a[8 + i] = (_Float16)p[16 + i];
    return a;
  };

  v8f acc[4] = {};
  v16h nB[4];
#pragma unroll
  for (int nt = 0; nt < 4; ++nt)
    nB[nt] = wmma_b_f16(vbase + (size_t)nt * 16 * L_, L_);
  v16h na = loadA(0);
  for (int kt = 0; kt < 16; ++kt) {
    const v16h ca = na;
    v16h cB[4];
#pragma unroll
    for (int nt = 0; nt < 4; ++nt) cB[nt] = nB[nt];
    if (kt + 1 < 16) {
      const _Float16* vb = vbase + (kt + 1) * 32;
#pragma unroll
      for (int nt = 0; nt < 4; ++nt)
        nB[nt] = wmma_b_f16(vb + (size_t)nt * 16 * L_, L_);
      na = loadA(kt + 1);
    }
#pragma unroll
    for (int nt = 0; nt < 4; ++nt) acc[nt] = wmma32(ca, cB[nt], acc[nt]);
  }
#pragma unroll
  for (int nt = 0; nt < 4; ++nt) {
    float* dst = out + ((size_t)b * L_ + q0 + mo) * D_ + nt * 16 + n;
#pragma unroll
    for (int r = 0; r < 8; ++r) dst[(size_t)r * D_] = acc[nt][r];
  }
}

// ---------------------------------------------------------------------------
// K3: out[b][qi][d] += sum_kj P[b][qi][kj] * edge_value[qi][kj][d]
// Per qi: GEMM M=b(64), N=d(64), K=kj(512); A = P f16, B = evT[qi][d][kj].
// Double-buffered one k-step ahead.  Grid: (L, B/16), block = 1 wave.
// ---------------------------------------------------------------------------
__global__ void __launch_bounds__(32)
edgev_kernel(const _Float16* __restrict__ P, const _Float16* __restrict__ evT,
             float* __restrict__ out) {
  const int qi   = blockIdx.x;
  const int b0   = blockIdx.y * 16;
  const int lane = threadIdx.x & 31;

  const _Float16* pBase = P + (((size_t)b0 * L_) + qi) * L_; // rows stride L*L
  const _Float16* eBase = evT + (size_t)qi * D_ * L_;        // [d][kj]

  v8f acc[4] = {};
  v16h nB[4];
#pragma unroll
  for (int nt = 0; nt < 4; ++nt)
    nB[nt] = wmma_b_f16(eBase + (size_t)nt * 16 * L_, L_);
  v16h na = wmma_a_f16(pBase, L_ * L_);

  for (int kt = 0; kt < 16; ++kt) {
    const v16h ca = na;
    v16h cB[4];
#pragma unroll
    for (int nt = 0; nt < 4; ++nt) cB[nt] = nB[nt];
    if (kt + 1 < 16) {
      const _Float16* eb = eBase + (kt + 1) * 32;
#pragma unroll
      for (int nt = 0; nt < 4; ++nt)
        nB[nt] = wmma_b_f16(eb + (size_t)nt * 16 * L_, L_);
      na = wmma_a_f16(pBase + (kt + 1) * 32, L_ * L_);
    }
#pragma unroll
    for (int nt = 0; nt < 4; ++nt) acc[nt] = wmma32(ca, cB[nt], acc[nt]);
  }

  const int n  = lane & 15;
  const int mo = (lane < 16) ? 0 : 8;
#pragma unroll
  for (int nt = 0; nt < 4; ++nt) {
    float* dst = out + ((size_t)(b0 + mo) * L_ + qi) * D_ + nt * 16 + n;
#pragma unroll
    for (int r = 0; r < 8; ++r) {
      float* d2 = dst + (size_t)r * L_ * D_;
      *d2 = *d2 + acc[nt][r];
    }
  }
}

// ---------------------------------------------------------------------------
// Launch. ws layout (bytes):
//   [0,   64Mi)  f32 bias[b][kj][qi]
//   [64,  96Mi)  f16 P[b][qi][kj]
//   [96, 100Mi)  f16 qh   [100,104Mi) f16 kh   [104,108Mi) f16 vT[b][d][kj]
//   [108,140Mi)  f16 ekh  [140,172Mi) f16 evT[qi][d][kj]
// Inputs (setup_inputs order): q, k, v, scale, edge_key, edge_value, edge_mask.
// ---------------------------------------------------------------------------
extern "C" void kernel_launch(void* const* d_in, const int* in_sizes, int n_in,
                              void* d_out, int out_size, void* d_ws,
                              size_t ws_size, hipStream_t stream) {
  (void)in_sizes; (void)n_in; (void)out_size; (void)ws_size;
  const float* q  = (const float*)d_in[0];
  const float* k  = (const float*)d_in[1];
  const float* v  = (const float*)d_in[2];
  const float* scp = (const float*)d_in[3];
  const float* ek = (const float*)d_in[4];
  const float* ev = (const float*)d_in[5];
  const unsigned char* mask = (const unsigned char*)d_in[6];
  float* out = (float*)d_out;

  char* ws = (char*)d_ws;
  const size_t nBLD = (size_t)B_ * L_ * D_;  // 2,097,152
  const size_t nLLD = (size_t)L_ * L_ * D_;  // 16,777,216
  const size_t nBLL = (size_t)B_ * L_ * L_;  // 16,777,216

  float*    bias = (float*)ws;                    ws += nBLL * 4;
  _Float16* P    = (_Float16*)ws;                 ws += nBLL * 2;
  _Float16* qh   = (_Float16*)ws;                 ws += nBLD * 2;
  _Float16* kh   = (_Float16*)ws;                 ws += nBLD * 2;
  _Float16* vT   = (_Float16*)ws;                 ws += nBLD * 2;
  _Float16* ekh  = (_Float16*)ws;                 ws += nLLD * 2;
  _Float16* evT  = (_Float16*)ws;

  // K0: precision/layout pre-pass (each element converted exactly once)
  cvt_kernel<<<(int)(nBLD / 2048), 256, 0, stream>>>(q, qh, (int)nBLD);
  cvt_kernel<<<(int)(nBLD / 2048), 256, 0, stream>>>(k, kh, (int)nBLD);
  cvt_kernel<<<(int)(nLLD / 2048), 256, 0, stream>>>(ek, ekh, (int)nLLD);
  transpose_cvt_kernel<<<dim3(L_ / 64, B_), 256, 0, stream>>>(v, vT, L_, D_);
  transpose_cvt_kernel<<<dim3(L_ / 64, L_), 256, 0, stream>>>(ev, evT, L_, D_);

  // K1..K3: all contractions on the WMMA path
  bias_kernel<<<dim3(L_, B_ / 16), 256, 0, stream>>>(qh, ekh, bias);
  attn_kernel<<<dim3(B_, L_ / 16), 32, 0, stream>>>(qh, kh, vT, scp, mask, bias,
                                                    P, out);
  edgev_kernel<<<dim3(L_, B_ / 16), 32, 0, stream>>>(P, evT, out);
}
